// Involution_18356690224020
// MI455X (gfx1250) — compile-verified, matrix operand order checked by reading, not run on hardware
//
#include <hip/hip_runtime.h>

// ---------------- problem sizes ----------------
#define BATCH 16
#define CIN   256
#define HH    56
#define WW    56
#define CRCH  128                 // reduced channels
#define GRP   16
#define KKG   144                 // 3*3*16 span channels
#define P2    (HH*WW)             // 3136
#define NPIX  (BATCH*P2)          // 50176 pixels (batch-major pixel index)

typedef __attribute__((ext_vector_type(16))) __bf16 v16bf;
typedef __attribute__((ext_vector_type(8)))  __bf16 v8bf;
typedef __attribute__((ext_vector_type(4)))  __bf16 v4bf;
typedef __attribute__((ext_vector_type(8)))  float  v8f;

// ---------------- workspace layout (bytes) ----------------
#define OFF_WRBF  0u            // 128*256 bf16  = 65536
#define OFF_WSBF  65536u        // 144*128 bf16  = 36864
#define OFF_STATS 102400u       // 256 f32 (sum,sumsq per cr)
#define OFF_SS    103424u       // 256 f32 (scale, shift)
#define OFF_T     104448u       // 128*50176 bf16 = 12845056
#define OFF_KERN  12949504u     // 144*50176 bf16 = 14450688
// total ~27.4 MB

__device__ __forceinline__ v16bf pack16(const __bf16* p) {
    v8bf lo = *(const v8bf*)p;
    v8bf hi = *(const v8bf*)(p + 16);
    v16bf r;
    #pragma unroll
    for (int i = 0; i < 8; i++) { r[i] = lo[i]; r[8 + i] = hi[i]; }
    return r;
}

// ---------------- kernel 0: weight convert + stats zero ----------------
__global__ void prep_kernel(const float* __restrict__ wr, const float* __restrict__ wsp,
                            __bf16* __restrict__ wrbf, __bf16* __restrict__ wsbf,
                            float* __restrict__ stats) {
    int i = blockIdx.x * blockDim.x + threadIdx.x;
    int stride = gridDim.x * blockDim.x;
    for (int idx = i; idx < 32768 + 18432 + 256; idx += stride) {
        if (idx < 32768)        wrbf[idx] = (__bf16)wr[idx];
        else if (idx < 51200)   wsbf[idx - 32768] = (__bf16)wsp[idx - 32768];
        else                    stats[idx - 51200] = 0.0f;
    }
}

// ---------------- kernel 1: reduce GEMM (t = x * Wr^T + b) + BN stats ----------------
// block = 256 threads = 8 waves; block covers 64 pixels x all 128 cr channels.
// Each wave register-blocks 4 N-subtiles; all 4 B tiles are loaded into distinct
// registers BEFORE the WMMA group so DScnt waits can stagger instead of serializing.
__global__ __launch_bounds__(256) void reduce_gemm_kernel(
        const float* __restrict__ x, const __bf16* __restrict__ wrbf,
        const float* __restrict__ b_reduce, __bf16* __restrict__ t,
        float* __restrict__ stats) {
    __shared__ __bf16 ldsXT[64 * 256];       // [pixel][channel], bf16, 32KB

    const int tid = threadIdx.x;
    const int p0  = blockIdx.x * 64;         // 3136 % 64 == 0 -> never crosses batch
    {   // stage: thread tid owns input channel tid, 64 consecutive pixels
        const int b  = p0 / P2;
        const int p2 = p0 - b * P2;
        const float4* src4 = (const float4*)(x + (long)(b * CIN + tid) * P2 + p2);
        #pragma unroll
        for (int i = 0; i < 16; i++) {
            float4 v4 = src4[i];
            ldsXT[(4 * i + 0) * 256 + tid] = (__bf16)v4.x;
            ldsXT[(4 * i + 1) * 256 + tid] = (__bf16)v4.y;
            ldsXT[(4 * i + 2) * 256 + tid] = (__bf16)v4.z;
            ldsXT[(4 * i + 3) * 256 + tid] = (__bf16)v4.w;
        }
        __builtin_prefetch(src4 + 16, 0, 0); // next pixel tile -> global_prefetch_b8
    }
    __syncthreads();

    const int wave   = tid >> 5;
    const int lane   = tid & 31;
    const int row    = lane & 15;            // A: M-row / B: N-col
    const int koff   = (lane < 16) ? 0 : 8;  // 16-bit operand K packing per ISA layout
    const int hi8    = koff;                 // D layout: lanes 16..31 hold M = r+8
    const int crbase = wave * 16;

    v8f acc0 = {}, acc1 = {}, acc2 = {}, acc3 = {};
    #pragma unroll
    for (int kk = 0; kk < 256; kk += 32) {
        const v16bf A   = pack16(wrbf + (long)(crbase + row) * 256 + kk + koff);
        const v16bf Bm0 = pack16(ldsXT + ( 0 + row) * 256 + kk + koff);
        const v16bf Bm1 = pack16(ldsXT + (16 + row) * 256 + kk + koff);
        const v16bf Bm2 = pack16(ldsXT + (32 + row) * 256 + kk + koff);
        const v16bf Bm3 = pack16(ldsXT + (48 + row) * 256 + kk + koff);
        acc0 = __builtin_amdgcn_wmma_f32_16x16x32_bf16(false, A, false, Bm0, (short)0, acc0, false, false);
        acc1 = __builtin_amdgcn_wmma_f32_16x16x32_bf16(false, A, false, Bm1, (short)0, acc1, false, false);
        acc2 = __builtin_amdgcn_wmma_f32_16x16x32_bf16(false, A, false, Bm2, (short)0, acc2, false, false);
        acc3 = __builtin_amdgcn_wmma_f32_16x16x32_bf16(false, A, false, Bm3, (short)0, acc3, false, false);
    }

    // epilogue: bias, store bf16 t (cr-major), accumulate BN sums
    const int pix = lane & 15;
    #pragma unroll
    for (int r = 0; r < 8; r++) {
        const int cr = crbase + r + hi8;
        const float bias = b_reduce[cr];
        float s = 0.0f, s2 = 0.0f;
        #pragma unroll
        for (int nt = 0; nt < 4; nt++) {
            float v = (nt == 0 ? acc0[r] : nt == 1 ? acc1[r] : nt == 2 ? acc2[r] : acc3[r]) + bias;
            t[(long)cr * NPIX + p0 + nt * 16 + pix] = (__bf16)v;
            s += v; s2 += v * v;
        }
        #pragma unroll
        for (int m = 1; m < 16; m <<= 1) {   // reduce across the 16-lane half
            s  += __shfl_xor(s,  m, 32);
            s2 += __shfl_xor(s2, m, 32);
        }
        if (pix == 0) {
            atomicAdd(&stats[2 * cr],     s);
            atomicAdd(&stats[2 * cr + 1], s2);
        }
    }
}

// ---------------- kernel 2: fold stats into BN scale/shift ----------------
__global__ void finalize_stats_kernel(const float* __restrict__ stats,
                                      const float* __restrict__ gamma,
                                      const float* __restrict__ beta,
                                      float* __restrict__ ss) {
    int c = threadIdx.x;                     // 128 threads
    const float inv_n = 1.0f / (float)NPIX;
    float mu  = stats[2 * c] * inv_n;
    float var = stats[2 * c + 1] * inv_n - mu * mu;
    float sc  = gamma[c] * rsqrtf(var + 1e-5f);
    ss[c]       = sc;
    ss[128 + c] = beta[c] - mu * sc;
}

// ---------------- kernel 3: span GEMM (kern = relu(bn(t)) * Ws^T + b) ----------------
// block = 288 threads = 9 waves; block covers 64 pixels x all 144 span channels.
__global__ __launch_bounds__(288) void span_gemm_kernel(
        const __bf16* __restrict__ t, const __bf16* __restrict__ wsbf,
        const float* __restrict__ ss, const float* __restrict__ b_span,
        __bf16* __restrict__ kern) {
    __shared__ __bf16 ldsT[64 * 128];        // [pixel][cr], bf16, 16KB

    const int tid = threadIdx.x;
    const int p0  = blockIdx.x * 64;
    for (int idx = tid; idx < 2048; idx += 288) {   // fused BN + ReLU during staging
        const int cr = idx >> 4;
        const int pc = (idx & 15) * 4;
        const float scl = ss[cr], shf = ss[128 + cr];
        v4bf tv = *(const v4bf*)(t + (long)cr * NPIX + p0 + pc);
        #pragma unroll
        for (int j = 0; j < 4; j++) {
            float v = (float)tv[j] * scl + shf;
            v = v > 0.0f ? v : 0.0f;
            ldsT[(pc + j) * 128 + cr] = (__bf16)v;
        }
    }
    __syncthreads();

    const int wave   = tid >> 5;             // 0..8 -> 9 * 16 = 144 channels
    const int lane   = tid & 31;
    const int row    = lane & 15;
    const int koff   = (lane < 16) ? 0 : 8;
    const int hi8    = koff;
    const int kobase = wave * 16;

    v8f acc0 = {}, acc1 = {}, acc2 = {}, acc3 = {};
    #pragma unroll
    for (int kk = 0; kk < 128; kk += 32) {
        const v16bf A   = pack16(wsbf + (long)(kobase + row) * 128 + kk + koff);
        const v16bf Bm0 = pack16(ldsT + ( 0 + row) * 128 + kk + koff);
        const v16bf Bm1 = pack16(ldsT + (16 + row) * 128 + kk + koff);
        const v16bf Bm2 = pack16(ldsT + (32 + row) * 128 + kk + koff);
        const v16bf Bm3 = pack16(ldsT + (48 + row) * 128 + kk + koff);
        acc0 = __builtin_amdgcn_wmma_f32_16x16x32_bf16(false, A, false, Bm0, (short)0, acc0, false, false);
        acc1 = __builtin_amdgcn_wmma_f32_16x16x32_bf16(false, A, false, Bm1, (short)0, acc1, false, false);
        acc2 = __builtin_amdgcn_wmma_f32_16x16x32_bf16(false, A, false, Bm2, (short)0, acc2, false, false);
        acc3 = __builtin_amdgcn_wmma_f32_16x16x32_bf16(false, A, false, Bm3, (short)0, acc3, false, false);
    }

    const int pix = lane & 15;
    #pragma unroll
    for (int r = 0; r < 8; r++) {
        const int ko = kobase + r + hi8;
        const float bias = b_span[ko];
        #pragma unroll
        for (int nt = 0; nt < 4; nt++) {
            float v = (nt == 0 ? acc0[r] : nt == 1 ? acc1[r] : nt == 2 ? acc2[r] : acc3[r]) + bias;
            kern[(long)ko * NPIX + p0 + nt * 16 + pix] = (__bf16)v;   // layout [x*16+g][pixel]
        }
    }
}

// ---------------- kernel 4: involution gather with async-to-LDS x staging ----------------
// block = (56,4): one (b,c) channel plane, 4 output rows; x halo tile (6 rows x 56)
// staged into LDS with gfx1250 GLOBAL_LOAD_ASYNC_TO_LDS_B32, completed by s_wait_asynccnt.
__global__ __launch_bounds__(224) void involution_kernel(
        const float* __restrict__ x, const __bf16* __restrict__ kern,
        float* __restrict__ out) {
    __shared__ float xt[6 * 56];                      // 1344 B halo tile

    const int w   = threadIdx.x;                      // 0..55 (coalesced)
    const int ty  = threadIdx.y;                      // 0..3
    const int h0  = blockIdx.x * 4;
    const int h   = h0 + ty;
    const int bc  = blockIdx.y;                       // b*256 + c
    const int b   = bc >> 8;
    const int c   = bc & 255;
    const int g   = c >> 4;
    const float* xc = x + (long)bc * P2;

    const int ftid = ty * 56 + w;                     // 0..223
    for (int idx = ftid; idx < 6 * 56; idx += 224) {
        const int r  = idx / 56;
        const int cw = idx - r * 56;
        const int gh = h0 - 1 + r;
        if (gh >= 0 && gh < HH) {
            unsigned lds_off = (unsigned)(size_t)(&xt[idx]);          // addr[31:0] = LDS offset
            unsigned long long ga = (unsigned long long)(xc + gh * WW + cw);
            asm volatile("global_load_async_to_lds_b32 %0, %1, off"
                         :: "v"(lds_off), "v"(ga) : "memory");
        } else {
            xt[idx] = 0.0f;                           // zero-fill halo rows outside image
        }
    }
    asm volatile("s_wait_asynccnt 0x0" ::: "memory"); // this wave's async copies done
    __syncthreads();                                  // tile visible to all waves

    const long pglob = (long)b * P2 + h * WW + w;
    float acc = 0.0f;
    #pragma unroll
    for (int ki = 0; ki < 3; ki++) {
        const int tr = ty + ki;                       // tile row for gh = h + ki - 1
        #pragma unroll
        for (int kj = 0; kj < 3; kj++) {
            const int wp = w + kj - 1;
            float kf = (float)kern[(long)((ki * 3 + kj) * GRP + g) * NPIX + pglob];
            float xv = (wp >= 0 && wp < WW) ? xt[tr * 56 + wp] : 0.0f;
            acc += xv * kf;
        }
    }
    out[(long)bc * P2 + h * WW + w] = acc;
}

// ---------------- host launch ----------------
extern "C" void kernel_launch(void* const* d_in, const int* in_sizes, int n_in,
                              void* d_out, int out_size, void* d_ws, size_t ws_size,
                              hipStream_t stream) {
    const float* x        = (const float*)d_in[0];
    const float* w_reduce = (const float*)d_in[1];
    const float* b_reduce = (const float*)d_in[2];
    const float* gamma    = (const float*)d_in[3];
    const float* beta     = (const float*)d_in[4];
    const float* w_span   = (const float*)d_in[5];
    const float* b_span   = (const float*)d_in[6];
    float* out            = (float*)d_out;

    char* ws = (char*)d_ws;
    __bf16* wrbf  = reinterpret_cast<__bf16*>(ws + OFF_WRBF);
    __bf16* wsbf  = reinterpret_cast<__bf16*>(ws + OFF_WSBF);
    float*  stats = reinterpret_cast<float*>(ws + OFF_STATS);
    float*  ss    = reinterpret_cast<float*>(ws + OFF_SS);
    __bf16* tbuf  = reinterpret_cast<__bf16*>(ws + OFF_T);
    __bf16* kern  = reinterpret_cast<__bf16*>(ws + OFF_KERN);

    prep_kernel<<<64, 256, 0, stream>>>(w_reduce, w_span, wrbf, wsbf, stats);
    reduce_gemm_kernel<<<NPIX / 64, 256, 0, stream>>>(x, wrbf, b_reduce, tbuf, stats);
    finalize_stats_kernel<<<1, 128, 0, stream>>>(stats, gamma, beta, ss);
    span_gemm_kernel<<<NPIX / 64, 288, 0, stream>>>(tbuf, wsbf, ss, b_span, kern);
    involution_kernel<<<dim3(HH / 4, BATCH * CIN), dim3(WW, 4), 0, stream>>>(x, kern, out);
}
